// MetaSpace_8022998909001
// MI455X (gfx1250) — matrix operand order repaired and checked by reading, not compiled, not run on hardware
//
#include <hip/hip_runtime.h>
#include <hip/hip_bf16.h>

typedef __attribute__((ext_vector_type(16))) _Float16 v16h;
typedef __attribute__((ext_vector_type(8)))  _Float16 v8h;
typedef __attribute__((ext_vector_type(8)))  float    v8f;

#define NB   32
#define NKP  17
#define CH   256
#define FH   96
#define FW   72
#define BN   544      /* NB*NKP  */
#define TOK  1088     /* BN*2    */

/* ---------------- Gaussian pooling: (B,C,H,W) gather -> (B*N, C) ---------- */
__global__ __launch_bounds__(256) void gpool_k(const float* __restrict__ fm,
                                               const float* __restrict__ kp,
                                               float* __restrict__ kptf) {
    int i = blockIdx.x;              // b*NKP + n
    int b = i / NKP;
    __shared__ float w[25];
    if (threadIdx.x < 25) {
        int dy = (int)threadIdx.x / 5 - 2, dx = (int)threadIdx.x % 5 - 2;
        w[threadIdx.x] = __expf(-(float)(dx * dx + dy * dy) * 0.125f); // 1/(2*sigma^2), sigma=2
    }
    __syncthreads();
    float kx = kp[i * 2 + 0] * ((float)FW / 288.0f);
    float ky = kp[i * 2 + 1] * ((float)FH / 384.0f);
    int x = (int)kx; x = x < 2 ? 2 : (x > FW - 3 ? FW - 3 : x);
    int y = (int)ky; y = y < 2 ? 2 : (y > FH - 3 ? FH - 3 : y);
    const float* base = fm + ((size_t)b * CH + threadIdx.x) * (FH * FW);
    float acc = 0.f, den = 0.f;
#pragma unroll
    for (int t = 0; t < 25; ++t) {
        int dy = t / 5 - 2, dx = t % 5 - 2;
        acc += w[t] * base[(y + dy) * FW + (x + dx)];
        den += w[t];
    }
    kptf[(size_t)i * CH + threadIdx.x] = acc / den;
}

/* -------- weight pack: (K,N) f32 row-major -> (N,K) f16 (transposed) ------ */
__global__ __launch_bounds__(256) void packWT_k(const float* __restrict__ in,
                                                _Float16* __restrict__ out, int K, int N) {
    int idx = blockIdx.x * 256 + threadIdx.x;
    if (idx >= K * N) return;
    int k = idx / N, n = idx % N;
    out[(size_t)n * K + k] = (_Float16)in[idx];
}

/* -------- combined = [kptf | meta] as (BN, 512) f16 ----------------------- */
__global__ __launch_bounds__(256) void packComb_k(const float* __restrict__ kptf,
                                                  const float* __restrict__ meta,
                                                  _Float16* __restrict__ out) {
    int idx = blockIdx.x * 256 + threadIdx.x;
    if (idx >= BN * 2 * CH) return;
    int i = idx >> 9, c = idx & 511;
    int n = i % NKP;
    float v = (c < CH) ? kptf[i * CH + c] : meta[n * CH + (c - CH)];
    out[idx] = (_Float16)v;
}

/* -------- stacked X: row 2i = kptf[i], row 2i+1 = meta[i%N]  (TOK,256) ---- */
__global__ __launch_bounds__(256) void packX_k(const float* __restrict__ kptf,
                                               const float* __restrict__ meta,
                                               _Float16* __restrict__ X) {
    int idx = blockIdx.x * 256 + threadIdx.x;
    if (idx >= BN * CH) return;
    int i = idx >> 8, c = idx & 255;
    X[(size_t)(2 * i) * CH + c]     = (_Float16)kptf[idx];
    X[(size_t)(2 * i + 1) * CH + c] = (_Float16)meta[(i % NKP) * CH + c];
}

/* ---------------- WMMA GEMM:  C(M,N) = A(M,K) * Bt(N,K)^T + bias ---------- */
/* wave32: each wave owns a 16x16 tile; block = 8 waves = 8 N-tiles.         */
__global__ __launch_bounds__(256) void gemm16_k(const _Float16* __restrict__ A,
                                                const _Float16* __restrict__ Bt,
                                                const float* __restrict__ bias,
                                                float* __restrict__ Cm,
                                                int M, int N, int K) {
    int lane = threadIdx.x & 31;
    int wv   = threadIdx.x >> 5;
    int mt   = blockIdx.x;
    int nt   = blockIdx.y * 8 + wv;
    if (mt * 16 >= M || nt * 16 >= N) return;
    int half = lane >> 4;            // 0: lanes 0-15, 1: lanes 16-31
    int lr   = lane & 15;
    const _Float16* arow = A  + (size_t)(mt * 16 + lr) * K;          // A row for this lane
    const _Float16* bcol = Bt + (size_t)(nt * 16 + lr) * K + half * 16; // B column (N-major)
    int aoff = half * 16;            // A layout: K = e + 16*half (matches B)
    v8f acc = {};
    for (int kk = 0; kk < K; kk += 32) {
        v16h a = *(const v16h*)(arow + kk + aoff);        // K = kk+16*half .. +15
        v16h bf = *(const v16h*)(bcol + kk);              // K = kk+16*half .. +15
        acc = __builtin_amdgcn_wmma_f32_16x16x32_f16(false, a, false, bf,
                                                     (short)0, acc, false, false);
    }
    float bv = bias[nt * 16 + lr];
    int r0 = mt * 16 + half * 8;     // D layout: M = r + 8*half, N = lr
#pragma unroll
    for (int r = 0; r < 8; ++r)
        Cm[(size_t)(r0 + r) * N + (nt * 16 + lr)] = acc[r] + bv;
}

/* ---------------- LayerNorm + ReLU over last dim (256) -> f16 ------------- */
__global__ __launch_bounds__(256) void lnrelu_k(const float* __restrict__ Hpre,
                                                const float* __restrict__ g,
                                                const float* __restrict__ bta,
                                                _Float16* __restrict__ Hh) {
    int row = blockIdx.x, c = threadIdx.x;
    float v = Hpre[(size_t)row * CH + c];
    __shared__ float red[CH];
    red[c] = v; __syncthreads();
    for (int s = 128; s > 0; s >>= 1) { if (c < s) red[c] += red[c + s]; __syncthreads(); }
    float mu = red[0] * (1.0f / CH);
    __syncthreads();
    float d = v - mu;
    red[c] = d * d; __syncthreads();
    for (int s = 128; s > 0; s >>= 1) { if (c < s) red[c] += red[c + s]; __syncthreads(); }
    float var = red[0] * (1.0f / CH);
    float y = d * rsqrtf(var + 1e-5f) * g[c] + bta[c];
    Hh[(size_t)row * CH + c] = (_Float16)fmaxf(y, 0.0f);
}

/* ---------------- gate = sigmoid(X @ Wg + bg)  (TOK, 8) ------------------- */
__global__ __launch_bounds__(256) void gate_k(const _Float16* __restrict__ X,
                                              const float* __restrict__ Wg,
                                              const float* __restrict__ bg,
                                              float* __restrict__ gate) {
    int idx = blockIdx.x * 256 + threadIdx.x;
    if (idx >= TOK * 8) return;
    int row = idx >> 3, h = idx & 7;
    float a = bg[h];
    for (int k = 0; k < CH; ++k)
        a += (float)X[(size_t)row * CH + k] * Wg[k * 8 + h];
    gate[idx] = 1.0f / (1.0f + __expf(-a));
}

/* ------- per-(m,head) attention over S=2 tokens; wave32 reductions -------- */
__global__ __launch_bounds__(256) void attn_k(const float* __restrict__ Q,
                                              const float* __restrict__ Km,
                                              const float* __restrict__ V,
                                              const float* __restrict__ gate,
                                              _Float16* __restrict__ Yg) {
    int m = blockIdx.x;
    int h = threadIdx.x >> 5;        // one wave per head (wave32, dk=32)
    int d = threadIdx.x & 31;
    size_t r0 = (size_t)(2 * m) * CH + h * 32 + d;
    size_t r1 = r0 + CH;
    float q0 = Q[r0], q1 = Q[r1], k0 = Km[r0], k1 = Km[r1], v0 = V[r0], v1 = V[r1];
    float p00 = q0 * k0, p01 = q0 * k1, p10 = q1 * k0, p11 = q1 * k1;
#pragma unroll
    for (int o = 16; o > 0; o >>= 1) {
        p00 += __shfl_xor(p00, o, 32);
        p01 += __shfl_xor(p01, o, 32);
        p10 += __shfl_xor(p10, o, 32);
        p11 += __shfl_xor(p11, o, 32);
    }
    const float sc = 0.17677669529663687f;   // 1/sqrt(32)
    p00 *= sc; p01 *= sc; p10 *= sc; p11 *= sc;
    float m0 = fmaxf(p00, p01), e00 = __expf(p00 - m0), e01 = __expf(p01 - m0);
    float m1 = fmaxf(p10, p11), e10 = __expf(p10 - m1), e11 = __expf(p11 - m1);
    float a00 = e00 / (e00 + e01), a01 = e01 / (e00 + e01);
    float a10 = e10 / (e10 + e11), a11 = e11 / (e10 + e11);
    float g0 = gate[(size_t)(2 * m) * 8 + h];
    float g1 = gate[(size_t)(2 * m + 1) * 8 + h];
    Yg[r0] = (_Float16)((a00 * v0 + a01 * v1) * g0);
    Yg[r1] = (_Float16)((a10 * v0 + a11 * v1) * g1);
}

/* -------- out = mean over 2 tokens of attended + projected ---------------- */
__global__ __launch_bounds__(256) void final_k(const float* __restrict__ att,
                                               const float* __restrict__ proj,
                                               float* __restrict__ out) {
    int idx = blockIdx.x * 256 + threadIdx.x;
    if (idx >= BN * CH) return;
    int i = idx >> 8, c = idx & 255;
    out[idx] = 0.5f * (att[(size_t)(2 * i) * CH + c] + att[(size_t)(2 * i + 1) * CH + c])
             + proj[idx];
}

extern "C" void kernel_launch(void* const* d_in, const int* in_sizes, int n_in,
                              void* d_out, int out_size, void* d_ws, size_t ws_size,
                              hipStream_t stream) {
    const float* fmap = (const float*)d_in[0];
    const float* kpts = (const float*)d_in[1];
    const float* meta = (const float*)d_in[2];
    const float* Wq = (const float*)d_in[3];  const float* bq = (const float*)d_in[4];
    const float* Wk = (const float*)d_in[5];  const float* bk = (const float*)d_in[6];
    const float* Wv = (const float*)d_in[7];  const float* bv = (const float*)d_in[8];
    const float* Wo = (const float*)d_in[9];  const float* bo = (const float*)d_in[10];
    const float* Wg = (const float*)d_in[11]; const float* bg = (const float*)d_in[12];
    const float* P1w = (const float*)d_in[13]; const float* P1b = (const float*)d_in[14];
    const float* ln_g = (const float*)d_in[15]; const float* ln_b = (const float*)d_in[16];
    const float* P2w = (const float*)d_in[17]; const float* P2b = (const float*)d_in[18];
    float* out = (float*)d_out;

    /* ---- workspace carve (f32 region then f16 region) ---- */
    float* wsf = (float*)d_ws;
    float* kptf = wsf;                 // 544*256
    float* Hpre = kptf + 139264;       // 544*256
    float* proj = Hpre + 139264;       // 544*256
    float* Qf   = proj + 139264;       // 1088*256
    float* Kf   = Qf   + 278528;
    float* Vf   = Kf   + 278528;
    float* att  = Vf   + 278528;       // 1088*256
    float* gate = att  + 278528;       // 1088*8
    _Float16* wsh = (_Float16*)(gate + 8704);
    _Float16* combH = wsh;             // 544*512
    _Float16* XH    = combH + 278528;  // 1088*256
    _Float16* Hh    = XH    + 278528;  // 544*256
    _Float16* YgH   = Hh    + 139264;  // 1088*256
    _Float16* P1wT  = YgH   + 278528;  // 256*512 (N-major)
    _Float16* P2wT  = P1wT  + 131072;  // 256*256
    _Float16* WqT   = P2wT  + 65536;
    _Float16* WkT   = WqT   + 65536;
    _Float16* WvT   = WkT   + 65536;
    _Float16* WoT   = WvT   + 65536;

    /* 1. gaussian pooling */
    gpool_k<<<BN, 256, 0, stream>>>(fmap, kpts, kptf);

    /* 2. weight packs (f32 -> f16, transposed to N-major) */
    packWT_k<<<(512 * 256 + 255) / 256, 256, 0, stream>>>(P1w, P1wT, 512, 256);
    packWT_k<<<(256 * 256 + 255) / 256, 256, 0, stream>>>(P2w, P2wT, 256, 256);
    packWT_k<<<(256 * 256 + 255) / 256, 256, 0, stream>>>(Wq, WqT, 256, 256);
    packWT_k<<<(256 * 256 + 255) / 256, 256, 0, stream>>>(Wk, WkT, 256, 256);
    packWT_k<<<(256 * 256 + 255) / 256, 256, 0, stream>>>(Wv, WvT, 256, 256);
    packWT_k<<<(256 * 256 + 255) / 256, 256, 0, stream>>>(Wo, WoT, 256, 256);

    /* 3. activation packs */
    packComb_k<<<(BN * 512 + 255) / 256, 256, 0, stream>>>(kptf, meta, combH);
    packX_k<<<(BN * 256 + 255) / 256, 256, 0, stream>>>(kptf, meta, XH);

    /* 4. MLP branch: P1 -> LN+ReLU -> P2 */
    gemm16_k<<<dim3(BN / 16, 2), 256, 0, stream>>>(combH, P1wT, P1b, Hpre, BN, 256, 512);
    lnrelu_k<<<BN, 256, 0, stream>>>(Hpre, ln_g, ln_b, Hh);
    gemm16_k<<<dim3(BN / 16, 2), 256, 0, stream>>>(Hh, P2wT, P2b, proj, BN, 256, 256);

    /* 5. MHA branch: QKV projections, gate, attention, output projection */
    gemm16_k<<<dim3(TOK / 16, 2), 256, 0, stream>>>(XH, WqT, bq, Qf, TOK, 256, 256);
    gemm16_k<<<dim3(TOK / 16, 2), 256, 0, stream>>>(XH, WkT, bk, Kf, TOK, 256, 256);
    gemm16_k<<<dim3(TOK / 16, 2), 256, 0, stream>>>(XH, WvT, bv, Vf, TOK, 256, 256);
    gate_k<<<(TOK * 8 + 255) / 256, 256, 0, stream>>>(XH, Wg, bg, gate);
    attn_k<<<BN, 256, 0, stream>>>(Qf, Kf, Vf, gate, YgH);
    gemm16_k<<<dim3(TOK / 16, 2), 256, 0, stream>>>(YgH, WoT, bo, att, TOK, 256, 256);

    /* 6. combine */
    final_k<<<(BN * 256 + 255) / 256, 256, 0, stream>>>(att, proj, out);
}
